// EPLL_59803124630064
// MI455X (gfx1250) — matrix operand order repaired and checked by reading, not compiled, not run on hardware
//
#include <hip/hip_runtime.h>
#include <math.h>

// ---------------- problem constants ----------------
#define HW      256
#define NPIX    (HW*HW)          // 65536
#define PS      6
#define D       36               // patch dim
#define NI      251              // patches per side
#define P_TOT   (NI*NI)          // 63001
#define K       200
#define KPAD    256              // 16 tiles of 16 comps -> 4 groups of 4 tiles
#define NGROUP  4                // component-tile groups per wave
#define GTILES  4                // tiles per group (N-blocking factor)
#define PHI_LD  1344             // 1296 (outer) + 36 (patch) + 1 (const) + 11 pad = 42*32
#define KSTEPS  (PHI_LD/32)      // 42
#define SIG2    0.01f

typedef __attribute__((ext_vector_type(8)))  float  v8f;
typedef __attribute__((ext_vector_type(16))) __bf16 v16bf;

union Frag16 { v16bf v; uint4 q[2]; };

__device__ __forceinline__ unsigned short f2bf(float x) {
    unsigned int u = __float_as_uint(x);
    u = (u + 0x7FFFu + ((u >> 16) & 1u)) >> 16;   // round-to-nearest-even
    return (unsigned short)u;
}

// ---------------------------------------------------------------------------
// Kernel 1: per-component prep. Builds (per beta):
//   Bmat[c][0..1295]    = bf16( vec(A_c) ),  A_c = U diag(1/(l+reg)) U^T
//   Bmat[c][1296..1331] = bf16( -2 * A_c mu_c )
//   Bmat[c][1332]       = bf16( -2 * (logw_c - 0.5 logdet - 0.5 mu A mu) )
//   Emat[c]             = f32 ( U diag(l/(l+reg)) U^T )    (Wiener filter)
// Padding components (c >= 200) get a huge score constant so argmin skips them.
// ---------------------------------------------------------------------------
__global__ __launch_bounds__(256)
void epll_prep(const float* __restrict__ mu,
               const float* __restrict__ logw,
               const float* __restrict__ eigvecs,   // [K][D][D]
               const float* __restrict__ eigvals,   // [K][D]
               float reg,                           // 1/beta
               float* __restrict__ Emat,            // [K][D][D]
               unsigned short* __restrict__ Bmat)   // [KPAD][PHI_LD] bf16
{
    int c   = blockIdx.x;
    int tid = threadIdx.x;

    if (c >= K) {  // padding component: all zeros, huge constant slot
        for (int q = tid; q < PHI_LD; q += 256) {
            unsigned short v = (q == 1296 + D) ? f2bf(1.0e30f) : (unsigned short)0;
            Bmat[c * PHI_LD + q] = v;
        }
        return;
    }

    __shared__ float Ush[D * D];
    __shared__ float Ash[D * D];
    __shared__ float sS[D], sT[D], sMu[D], sAmu[D], sLreg[D];

    for (int q = tid; q < D * D; q += 256) Ush[q] = eigvecs[c * D * D + q];
    if (tid < D) {
        float ev = eigvals[c * D + tid];
        float l  = ev + reg;
        sLreg[tid] = l;
        sS[tid]    = 1.0f / l;
        sT[tid]    = ev / l;
        sMu[tid]   = mu[c * D + tid];
    }
    __syncthreads();

    for (int q = tid; q < D * D; q += 256) {
        int d = q / D, f = q % D;
        float a = 0.f, e2 = 0.f;
        #pragma unroll 6
        for (int e = 0; e < D; ++e) {
            float p = Ush[d * D + e] * Ush[f * D + e];
            a  += p * sS[e];
            e2 += p * sT[e];
        }
        Ash[q] = a;
        Bmat[c * PHI_LD + q] = f2bf(a);
        Emat[c * D * D + q]  = e2;
    }
    __syncthreads();

    if (tid < D) {
        float am = 0.f;
        #pragma unroll 6
        for (int f = 0; f < D; ++f) am += Ash[tid * D + f] * sMu[f];
        sAmu[tid] = am;
        Bmat[c * PHI_LD + 1296 + tid] = f2bf(-2.0f * am);
    }
    __syncthreads();

    if (tid == 0) {
        float muAmu = 0.f, logdet = 0.f;
        #pragma unroll 6
        for (int d = 0; d < D; ++d) { muAmu += sMu[d] * sAmu[d]; logdet += logf(sLreg[d]); }
        float cst = logw[c] - 0.5f * logdet - 0.5f * muAmu;
        Bmat[c * PHI_LD + 1296 + D] = f2bf(-2.0f * cst);
    }
    if (tid >= 1 && tid < PHI_LD - (1296 + D))   // zero slots 1333..1343
        Bmat[c * PHI_LD + 1296 + D + tid] = 0;
}

// ---------------------------------------------------------------------------
// Kernel 2: main routing + Wiener. One wave (32 threads) per 16-patch tile.
//  - gather 16 patches of x into LDS (f32)
//  - build augmented phi tile [16 x 1344] in bf16 in LDS
//  - 4 groups x (42 k-steps, 4 N-blocked accumulators): each A fragment is
//    reused by 4 v_wmma_f32_16x16x32_bf16 -> 0.5 ds_load per WMMA
//    score = xAx - 2*cross - 2*const  -> per-lane argmin, LDS cross-lane reduce
//  - f32 Wiener matvec with E[k*] -> est buffer (deterministic overlap-add later)
// ---------------------------------------------------------------------------
__global__ __launch_bounds__(32)
void epll_main(const float* __restrict__ xin,
               const float* __restrict__ Emat,
               const unsigned short* __restrict__ Bmat,
               float* __restrict__ est)             // [P_TOT][D]
{
    __shared__ unsigned short phi[16 * PHI_LD];     // 42 KB
    __shared__ float patf[16 * D];
    __shared__ float sc_s[16 * 16];
    __shared__ int   sc_c[16 * 16];
    __shared__ int   kstar[16];

    const int lane = threadIdx.x;
    const int base = blockIdx.x * 16;

    // ---- gather 16 patches ----
    for (int idx = lane; idx < 16 * D; idx += 32) {
        int tp = idx / D, d = idx % D;
        int p = base + tp;
        float v = 0.f;
        if (p < P_TOT) {
            int i = p / NI, j = p % NI;
            v = xin[(i + d / PS) * HW + (j + d % PS)];
        }
        patf[idx] = v;
    }
    __syncthreads();

    // ---- build bf16 phi tile: [outer(patch,patch) | patch | 1 | 0-pad] ----
    for (int tp = 0; tp < 16; ++tp) {
        const float* pr = &patf[tp * D];
        for (int q = lane; q < PHI_LD; q += 32) {
            float v;
            if (q < 1296) {
                int a = q / D;
                v = pr[a] * pr[q - a * D];
            } else if (q < 1296 + D) v = pr[q - 1296];
            else if (q == 1296 + D)  v = 1.0f;
            else                     v = 0.f;
            phi[tp * PHI_LD + q] = f2bf(v);
        }
    }
    __syncthreads();

    const int h  = lane >> 4;    // lane half
    const int ln = lane & 15;

    float best[8];
    int   bestc[8];
    #pragma unroll
    for (int r = 0; r < 8; ++r) { best[r] = 3.0e37f; bestc[r] = 0; }

    // A fragment: lane holds row M=ln, K in {k0+8h..+7} u {k0+16+8h..+7}
    const unsigned short* arow = phi + ln * PHI_LD + 8 * h;

    for (int g = 0; g < NGROUP; ++g) {
        // B fragments: lane holds column c = g*64 + tt*16 + ln,
        // K in {k0+8h..+7} u {k0+16+8h..+7}
        const unsigned short* b0 = Bmat + (g * (GTILES * 16) + ln) * PHI_LD + 8 * h;

        if (g + 1 < NGROUP)   // prefetch next group's B columns into cache
            __builtin_prefetch(b0 + (GTILES * 16) * PHI_LD, 0, 1);

        v8f acc[GTILES];
        #pragma unroll
        for (int tt = 0; tt < GTILES; ++tt)
            acc[tt] = (v8f){0.f, 0.f, 0.f, 0.f, 0.f, 0.f, 0.f, 0.f};

        for (int ks = 0; ks < KSTEPS; ++ks) {
            const int k0 = ks * 32;
            Frag16 fa;
            fa.q[0] = *(const uint4*)(arow + k0);
            fa.q[1] = *(const uint4*)(arow + k0 + 16);
            Frag16 fb[GTILES];
            #pragma unroll
            for (int tt = 0; tt < GTILES; ++tt) {
                const unsigned short* bc = b0 + tt * 16 * PHI_LD + k0;
                fb[tt].q[0] = *(const uint4*)(bc);
                fb[tt].q[1] = *(const uint4*)(bc + 16);
            }
            #pragma unroll
            for (int tt = 0; tt < GTILES; ++tt) {
                acc[tt] = __builtin_amdgcn_wmma_f32_16x16x32_bf16(
                              false, fa.v, false, fb[tt].v, (short)0, acc[tt],
                              false, false);
            }
        }
        #pragma unroll
        for (int tt = 0; tt < GTILES; ++tt) {
            int comp = g * (GTILES * 16) + tt * 16 + ln;
            #pragma unroll
            for (int r = 0; r < 8; ++r) {    // lane covers rows M = 8h + r
                if (acc[tt][r] < best[r]) { best[r] = acc[tt][r]; bestc[r] = comp; }
            }
        }
    }

    // ---- cross-lane argmin per patch ----
    #pragma unroll
    for (int r = 0; r < 8; ++r) {
        sc_s[(8 * h + r) * 16 + ln] = best[r];
        sc_c[(8 * h + r) * 16 + ln] = bestc[r];
    }
    __syncthreads();
    if (lane < 16) {
        float bs = sc_s[lane * 16];
        int   bc = sc_c[lane * 16];
        for (int q = 1; q < 16; ++q) {
            float s = sc_s[lane * 16 + q];
            if (s < bs) { bs = s; bc = sc_c[lane * 16 + q]; }
        }
        kstar[lane] = bc;
    }
    __syncthreads();

    // ---- routed Wiener filter: est[p] = E[k*] patch[p] (f32, float4 loads) ----
    for (int o = lane; o < 16 * D; o += 32) {
        int tp = o / D, d = o % D;
        int p = base + tp;
        if (p >= P_TOT) continue;
        const float4* Erow = (const float4*)(Emat + kstar[tp] * (D * D) + d * D);
        const float4* pv   = (const float4*)(&patf[tp * D]);
        float e = 0.f;
        #pragma unroll
        for (int f = 0; f < D / 4; ++f) {
            float4 er = Erow[f];
            float4 pw = pv[f];
            e += er.x * pw.x + er.y * pw.y + er.z * pw.z + er.w * pw.w;
        }
        est[p * D + d] = e;
    }
}

// ---------------------------------------------------------------------------
// Kernel 3: deterministic overlap-add (gather form) + data-fidelity step.
//   x_tilde[pix] = mean of est over covering patches; x = (y + b*x_tilde)/(1+b)
//   (beta * SIGMA_SQ == b exactly for b in {1,4,8,16,32})
// ---------------------------------------------------------------------------
__global__ __launch_bounds__(256)
void epll_combine(const float* __restrict__ y,
                  const float* __restrict__ est,
                  float b,
                  float* __restrict__ xout)
{
    int idx = blockIdx.x * 256 + threadIdx.x;
    if (idx >= NPIX) return;
    int i = idx >> 8, j = idx & 255;
    float acc = 0.f;
    int cnt = 0;
    #pragma unroll
    for (int di = 0; di < PS; ++di) {
        int i0 = i - di;
        if (i0 < 0 || i0 > NI - 1) continue;
        #pragma unroll
        for (int dj = 0; dj < PS; ++dj) {
            int j0 = j - dj;
            if (j0 < 0 || j0 > NI - 1) continue;
            acc += est[(i0 * NI + j0) * D + di * PS + dj];
            ++cnt;
        }
    }
    float xt = acc / (float)cnt;
    xout[idx] = (y[idx] + b * xt) / (1.0f + b);
}

// ---------------------------------------------------------------------------
extern "C" void kernel_launch(void* const* d_in, const int* in_sizes, int n_in,
                              void* d_out, int out_size, void* d_ws, size_t ws_size,
                              hipStream_t stream)
{
    (void)in_sizes; (void)n_in; (void)out_size; (void)ws_size;
    const float* y       = (const float*)d_in[0];
    const float* mu      = (const float*)d_in[1];
    const float* logw    = (const float*)d_in[2];
    const float* eigvecs = (const float*)d_in[3];
    const float* eigvals = (const float*)d_in[4];
    float* out = (float*)d_out;

    // workspace layout (~11 MB)
    float* x_ws = (float*)d_ws;                       // NPIX f32
    float* Emat = x_ws + NPIX;                        // K*D*D f32
    float* estb = Emat + K * D * D;                   // P_TOT*D f32
    unsigned short* Bmat = (unsigned short*)(estb + P_TOT * D); // KPAD*PHI_LD bf16

    const float bvals[5] = {1.f, 4.f, 8.f, 16.f, 32.f};
    const int n_patch_blocks = (P_TOT + 15) / 16;     // 3938

    const float* xin = y;                             // x_init = A^T y = y
    for (int it = 0; it < 5; ++it) {
        float beta = bvals[it] / SIG2;
        float reg  = 1.0f / beta;
        epll_prep<<<KPAD, 256, 0, stream>>>(mu, logw, eigvecs, eigvals, reg, Emat, Bmat);
        epll_main<<<n_patch_blocks, 32, 0, stream>>>(xin, Emat, Bmat, estb);
        float* xout = (it == 4) ? out : x_ws;
        epll_combine<<<(NPIX + 255) / 256, 256, 0, stream>>>(y, estb, bvals[it], xout);
        xin = x_ws;
    }
}
